// SimpleMoE_86406152061637
// MI455X (gfx1250) — matrix operand order repaired and checked by reading, not compiled, not run on hardware
//
#include <hip/hip_runtime.h>
#include <math.h>

// Problem constants (match reference)
#define Hd   1024
#define Fd   4096
#define Ed   8
#define Td   8192      // 4 * 2048 tokens
#define TILE 16        // tokens per FFN workgroup
#define FC   128       // F-chunk width (8 waves * 16)

typedef __bf16 bf16_t;
typedef bf16_t v16bf __attribute__((ext_vector_type(16)));
typedef float  v8f   __attribute__((ext_vector_type(8)));
typedef unsigned int uint4v __attribute__((ext_vector_type(4)));

__device__ __forceinline__ unsigned short f2bf(float f) {
    unsigned int u = __float_as_uint(f);
    u += 0x7FFFu + ((u >> 16) & 1u);          // round-to-nearest-even
    return (unsigned short)(u >> 16);
}

union FragB16 {
    v16bf v;
    unsigned short us[16];
    uint4v q[2];
};

// 16x32 bf16 A-fragment from an LDS row (ISA 16-bit A layout:
// lane M = lane&15; elems 0..7 -> K = aoff+0..7, 8..15 -> K = aoff+16..23,
// aoff = (lane>=16)*8). Two 16B LDS chunks per lane.
__device__ __forceinline__ v16bf load_a_lds(const unsigned short* row, int k0) {
    FragB16 f;
    f.q[0] = *(const uint4v*)(row + k0);
    f.q[1] = *(const uint4v*)(row + k0 + 16);
    return f.v;
}

// Pre-swizzled B-fragment: [fragIdx][lane][16 bf16] -> two coalesced b128 loads.
__device__ __forceinline__ v16bf load_b_swz(const unsigned short* base) {
    FragB16 f;
    const uint4v* p = (const uint4v*)base;
    f.q[0] = p[0];
    f.q[1] = p[1];
    return f.v;
}

// ---------------- kernel 0: zero output + counters ----------------
__global__ void moe_zero(float* __restrict__ out, int n, int* __restrict__ counts) {
    int i = blockIdx.x * blockDim.x + threadIdx.x;
    if (i < n) out[i] = 0.0f;
    if (i < 16) counts[i] = 0;
}

// ---------------- kernel 1a/1b: swizzle weights into WMMA B-fragment order ----
// B-fragment layout (K=32, N=16, bf16): lane holds N = lane&15,
// element j -> K = (lane>>4)*8 + (j<8 ? j : j+8)  (matches A-fragment K order).
// One thread produces one lane's 16 bf16 (32B) -> fully coalesced stores.
__global__ void moe_swz_w1(const float* __restrict__ w1, unsigned short* __restrict__ w1s) {
    const int id   = blockIdx.x * 256 + threadIdx.x;      // [0, Ed*32*256*32)
    const int lane = id & 31;
    const int t    = id >> 5;                             // fragment index
    const int ntile = t % (Fd / 16);
    const int t2    = t / (Fd / 16);
    const int kt    = t2 % (Hd / 32);
    const int e     = t2 / (Hd / 32);
    const float* src = w1 + (size_t)e * Hd * Fd
                     + (size_t)(kt * 32 + (lane >> 4) * 8) * Fd
                     + (ntile * 16 + (lane & 15));
    FragB16 f;
#pragma unroll
    for (int j = 0; j < 16; ++j) f.us[j] = f2bf(src[(size_t)(j < 8 ? j : j + 8) * Fd]);
    uint4v* dst = (uint4v*)(w1s + (size_t)id * 16);
    dst[0] = f.q[0];
    dst[1] = f.q[1];
}

__global__ void moe_swz_w2(const float* __restrict__ w2, unsigned short* __restrict__ w2s) {
    const int id   = blockIdx.x * 256 + threadIdx.x;      // [0, Ed*128*64*32)
    const int lane = id & 31;
    const int t    = id >> 5;
    const int ntile = t % (Hd / 16);
    const int t2    = t / (Hd / 16);
    const int kt    = t2 % (Fd / 32);
    const int e     = t2 / (Fd / 32);
    const float* src = w2 + (size_t)e * Fd * Hd
                     + (size_t)(kt * 32 + (lane >> 4) * 8) * Hd
                     + (ntile * 16 + (lane & 15));
    FragB16 f;
#pragma unroll
    for (int j = 0; j < 16; ++j) f.us[j] = f2bf(src[(size_t)(j < 8 ? j : j + 8) * Hd]);
    uint4v* dst = (uint4v*)(w2s + (size_t)id * 16);
    dst[0] = f.q[0];
    dst[1] = f.q[1];
}

// ---------------- kernel 2: router (one wave32 per token) ----------------
__global__ void moe_router(const float* __restrict__ x,
                           const float* __restrict__ rw,
                           int* __restrict__ counts,
                           int* __restrict__ tok,
                           float* __restrict__ wt) {
    __shared__ float s_rw[Ed * Hd];            // 32 KB
    const int tid = threadIdx.x;
    for (int i = tid; i < Ed * Hd; i += 256) s_rw[i] = rw[i];
    __syncthreads();

    const int wave = tid >> 5;
    const int lane = tid & 31;
    const int t = blockIdx.x * 8 + wave;
    const float* xr = x + (size_t)t * Hd;

    float acc[Ed];
#pragma unroll
    for (int e = 0; e < Ed; ++e) acc[e] = 0.0f;
    for (int h = lane; h < Hd; h += 32) {
        float xv = xr[h];
#pragma unroll
        for (int e = 0; e < Ed; ++e) acc[e] += xv * s_rw[e * Hd + h];
    }
#pragma unroll
    for (int off = 16; off > 0; off >>= 1) {
#pragma unroll
        for (int e = 0; e < Ed; ++e) acc[e] += __shfl_xor(acc[e], off, 32);
    }
    if (lane == 0) {
        int i0 = 0; float l0 = acc[0];
#pragma unroll
        for (int e = 1; e < Ed; ++e) if (acc[e] > l0) { l0 = acc[e]; i0 = e; }
        int i1 = -1; float l1 = -3.4e38f;
#pragma unroll
        for (int e = 0; e < Ed; ++e) if (e != i0 && acc[e] > l1) { l1 = acc[e]; i1 = e; }
        float p0 = 1.0f / (1.0f + __expf(l1 - l0));   // renormalized top-2 softmax
        float p1 = 1.0f - p0;
        int pos0 = atomicAdd(&counts[i0], 1);
        tok[i0 * Td + pos0] = t; wt[i0 * Td + pos0] = p0;
        int pos1 = atomicAdd(&counts[i1], 1);
        tok[i1 * Td + pos1] = t; wt[i1 * Td + pos1] = p1;
    }
}

// ---------------- kernel 3: fused expert FFN (WMMA bf16) ----------------
__global__ void __launch_bounds__(256, 1)
moe_ffn(const float* __restrict__ x,
        const unsigned short* __restrict__ w1s,
        const float* __restrict__ b1,
        const unsigned short* __restrict__ w2s,
        const float* __restrict__ b2,
        const int* __restrict__ counts,
        const int* __restrict__ tok,
        const float* __restrict__ wt,
        float* __restrict__ out) {
    const int TILES = Td / TILE;
    const int e    = blockIdx.x / TILES;
    const int tile = blockIdx.x % TILES;
    const int base = tile * TILE;
    const int n_e  = counts[e];
    if (base >= n_e) return;
    const int rows = (n_e - base < TILE) ? (n_e - base) : TILE;

    constexpr int XS = Hd + 8;   // 1032 bf16: 16B-aligned, bank-staggered rows
    constexpr int HS = FC + 8;   // 136 bf16
    __shared__ unsigned short lds_x[TILE * XS];   // 33.0 KB
    __shared__ unsigned short lds_h[TILE * HS];   //  4.3 KB
    __shared__ int   s_tok[TILE];
    __shared__ float s_wt[TILE];

    const int tid = threadIdx.x;
    if (tid < TILE) {
        int rr = (tid < rows) ? tid : 0;
        s_tok[tid] = tok[e * Td + base + rr];
        s_wt[tid]  = (tid < rows) ? wt[e * Td + base + tid] : 0.0f;
    }
    __syncthreads();

    // stage gathered X tile as bf16 (coalesced f32 reads; one-time cost)
    for (int i = tid; i < TILE * Hd; i += 256) {
        int r = i >> 10, c = i & (Hd - 1);
        lds_x[r * XS + c] = f2bf(x[(size_t)s_tok[r] * Hd + c]);
    }
    __syncthreads();

    const float* b1e = b1 + e * Fd;
    const float* b2e = b2 + e * Hd;

    const int wv   = tid >> 5;
    const int lane = tid & 31;
    const int nlo  = lane & 15;
    const int aoff = (lane >> 4) * 8;    // A-frag K offset / C-D row offset

    v8f acc[8];
#pragma unroll
    for (int nt = 0; nt < 8; ++nt) acc[nt] = (v8f)(0.0f);

    const unsigned short* xrow = lds_x + nlo * XS;
    const unsigned short* hrow = lds_h + nlo * HS;
    const int c0 = wv * FC;              // this wave's output column slice

    // per-expert bases into pre-swizzled weight images
    const unsigned short* w1e = w1s + (size_t)e * (Hd / 32) * (Fd / 16) * 512;
    const unsigned short* w2e = w2s + (size_t)e * (Fd / 32) * (Hd / 16) * 512;

    for (int fc = 0; fc < Fd; fc += FC) {
        // ---- Phase A: h[:, fc+wv*16 .. +16) = gelu(X @ W1 + b1) ----
        const int n0 = fc + wv * 16;
        const int nt1 = n0 >> 4;
        v8f c1 = (v8f)(0.0f);
        for (int kk = 0; kk < Hd; kk += 32) {
            v16bf a = load_a_lds(xrow, kk + aoff);
            const unsigned short* bp =
                w1e + ((size_t)((kk >> 5) * (Fd / 16) + nt1) * 32 + lane) * 16;
            __builtin_prefetch(bp + (size_t)(Fd / 16) * 512, 0, 1); // next k-step
            v16bf b = load_b_swz(bp);
            c1 = __builtin_amdgcn_wmma_f32_16x16x32_bf16(
                     false, a, false, b, (short)0, c1, false, false);
        }
        const float bias = b1e[n0 + nlo];
#pragma unroll
        for (int r = 0; r < 8; ++r) {
            float v = c1[r] + bias;
            v = 0.5f * v * (1.0f + erff(v * 0.70710678118654752f)); // exact GELU
            lds_h[(r + aoff) * HS + (n0 - fc) + nlo] = f2bf(v);
        }
        __syncthreads();

        // ---- Phase B: acc[:, c0..c0+128) += h_chunk @ W2[fc..fc+128, :] ----
        v16bf af[4];
#pragma unroll
        for (int q = 0; q < 4; ++q) af[q] = load_a_lds(hrow, q * 32 + aoff);
#pragma unroll
        for (int nt = 0; nt < 8; ++nt) {
            const int ntile2 = wv * 8 + nt;
#pragma unroll
            for (int q = 0; q < 4; ++q) {
                const unsigned short* bp =
                    w2e + ((size_t)((((fc >> 5) + q)) * (Hd / 16) + ntile2) * 32 + lane) * 16;
                v16bf b = load_b_swz(bp);
                acc[nt] = __builtin_amdgcn_wmma_f32_16x16x32_bf16(
                              false, af[q], false, b, (short)0, acc[nt], false, false);
            }
        }
        __syncthreads();
    }

    // ---- epilogue: out[t, col] += p * (acc + b2) (exactly 2 adders/elem,
    //      fp add commutative -> bitwise deterministic) ----
#pragma unroll
    for (int nt = 0; nt < 8; ++nt) {
        const int col = c0 + nt * 16 + nlo;
        const float b2v = b2e[col];
#pragma unroll
        for (int r = 0; r < 8; ++r) {
            const int row = r + aoff;
            if (row < rows) {
                float yv = acc[nt][r] + b2v;
                atomicAdd(&out[(size_t)s_tok[row] * Hd + col], s_wt[row] * yv);
            }
        }
    }
}

extern "C" void kernel_launch(void* const* d_in, const int* in_sizes, int n_in,
                              void* d_out, int out_size, void* d_ws, size_t ws_size,
                              hipStream_t stream) {
    const float* x  = (const float*)d_in[0];
    const float* rw = (const float*)d_in[1];
    const float* w1 = (const float*)d_in[2];
    const float* b1 = (const float*)d_in[3];
    const float* w2 = (const float*)d_in[4];
    const float* b2 = (const float*)d_in[5];
    float* out = (float*)d_out;

    // workspace layout
    char* ws = (char*)d_ws;
    int*   counts = (int*)ws;                              // 16 ints
    int*   tok    = (int*)(ws + 64);                       // Ed*Td ints  (256 KB)
    float* wtp    = (float*)(ws + 64 + Ed * Td * 4);       // Ed*Td f32   (256 KB)
    const size_t OFF_W1S = 1u << 20;                       // 1 MB
    const size_t W1S_BYTES = (size_t)Ed * Hd * Fd * 2;     // 64 MB
    unsigned short* w1s = (unsigned short*)(ws + OFF_W1S);
    unsigned short* w2s = (unsigned short*)(ws + OFF_W1S + W1S_BYTES);

    const int n_out = Td * Hd;
    moe_zero<<<(n_out + 255) / 256, 256, 0, stream>>>(out, n_out, counts);

    const int n_sw1 = Ed * (Hd / 32) * (Fd / 16) * 32;     // threads (one per lane-frag)
    const int n_sw2 = Ed * (Fd / 32) * (Hd / 16) * 32;
    moe_swz_w1<<<n_sw1 / 256, 256, 0, stream>>>(w1, w1s);
    moe_swz_w2<<<n_sw2 / 256, 256, 0, stream>>>(w2, w2s);

    moe_router<<<Td / 8, 256, 0, stream>>>(x, rw, counts, tok, wtp);
    moe_ffn<<<Ed * (Td / TILE), 256, 0, stream>>>(x, w1s, b1, w2s, b2,
                                                  counts, tok, wtp, out);
}
